// Attention_30356828848204
// MI455X (gfx1250) — matrix operand order repaired and checked by reading, not compiled, not run on hardware
//
#include <hip/hip_runtime.h>
#include <hip/hip_bf16.h>

#define BATCH 8
#define NSEQ  4096
#define DDIM  128
#define L2E   1.44269504088896340736f
#define TROWS 128
#define TSTRIDE 136                     // padded halves per row in LDS
#define TELEMS (TROWS * TSTRIDE)

typedef __attribute__((ext_vector_type(16))) __bf16 v16bf;
typedef __attribute__((ext_vector_type(8)))  __bf16 v8bf;
typedef __attribute__((ext_vector_type(8)))  float  v8f;
typedef __attribute__((ext_vector_type(4)))  unsigned int u32x4;
typedef __attribute__((ext_vector_type(8)))  int i32x8;
typedef __attribute__((ext_vector_type(4)))  int i32x4;

// Build a 16-half fragment from two 8-half (16B) aligned chunks.
__device__ __forceinline__ v16bf make_frag(const __bf16* p, int off0, int off1) {
  v8bf lo = *(const v8bf*)(p + off0);
  v8bf hi = *(const v8bf*)(p + off1);
  v16bf r;
#pragma unroll
  for (int i = 0; i < 8; ++i) { r[i] = lo[i]; r[i + 8] = hi[i]; }
  return r;
}

// Tensor Data Mover: DMA one 128x128 bf16 tile (row stride 128 elems in global)
// into LDS at lds_addr, padding each 256B row with 16B (-> 136-half stride).
__device__ __forceinline__ void tdm_load_tile(const __bf16* gsrc, unsigned lds_addr) {
  unsigned long long ga = (unsigned long long)gsrc;
  u32x4 g0;
  g0[0] = 1u;                                   // count=1 (valid), user mode
  g0[1] = lds_addr;                             // LDS byte address
  g0[2] = (unsigned)ga;                         // global_addr[31:0]
  g0[3] = (unsigned)((ga >> 32) & 0x1FFFFFFu)   // global_addr[56:32]
          | (2u << 30);                         // type=2 ("image")
  i32x8 g1;
  g1[0] = (1 << 16)                             // data_size = 2 bytes
        | (1 << 20)                             // pad_enable
        | (5 << 22)                             // pad_interval: 64 DWORDs (256B)
        | (3 << 25);                            // pad_amount: 4 DWORDs (16B)
  g1[1] = (int)(128u << 16);                    // tensor_dim0 = 128
  g1[2] = (int)(128u << 16);                    // tensor_dim1 = 128
  g1[3] = (int)(128u << 16);                    // tile_dim0 = 128
  g1[4] = 128;                                  // tile_dim1 = 128, tile_dim2 = 0
  g1[5] = 128;                                  // tensor_dim0_stride = 128
  g1[6] = 0;
  g1[7] = 0;
  i32x4 z4 = {0, 0, 0, 0};
#if defined(__clang_major__) && __clang_major__ >= 23
  i32x8 z8 = {0, 0, 0, 0, 0, 0, 0, 0};
  __builtin_amdgcn_tensor_load_to_lds(g0, g1, z4, z4, z8, 0);
#else
  __builtin_amdgcn_tensor_load_to_lds(g0, g1, z4, z4, 0);
#endif
}

// ---------------------------------------------------------------- kernel 0
__global__ void cvt_kernel(const float* __restrict__ x, __bf16* __restrict__ xb, int n) {
  int i = blockIdx.x * blockDim.x + threadIdx.x;
  int stride = gridDim.x * blockDim.x;
  for (; i < n; i += stride) xb[i] = (__bf16)x[i];
}

// ---------------------------------------------------------------- kernel 1
// Per-row softmax stats over the full score matrix (online max/sum).
// Double-buffered TDM pipeline: wave 0 DMAs key tile t+1 while all compute on t.
__global__ __launch_bounds__(256) void stats_kernel(const __bf16* __restrict__ xb,
                                                    float* __restrict__ mrow,
                                                    float* __restrict__ zinv) {
  const int b     = blockIdx.x >> 5;
  const int qt    = blockIdx.x & 31;
  const int qbase = qt * 128;
  const int wave  = threadIdx.x >> 5;
  const int lane  = threadIdx.x & 31;
  const int h     = lane >> 4;
  const int lr    = lane & 15;

  __shared__ __attribute__((aligned(16))) __bf16 kt[2 * TELEMS];
  const unsigned ktaddr = (unsigned)(unsigned long long)(&kt[0]);

  // A-matrix fragments: this wave's 16 query rows, K = full D (4 chunks of 32).
  const __bf16* qrow = xb + ((size_t)b * NSEQ + qbase + wave * 16 + lr) * DDIM;
  v16bf afr[4];
#pragma unroll
  for (int c = 0; c < 4; ++c)
    afr[c] = make_frag(qrow, 32 * c + 8 * h, 32 * c + 16 + 8 * h);

  float m[8], z[8];
#pragma unroll
  for (int v = 0; v < 8; ++v) { m[v] = -1e30f; z[v] = 0.f; }

  if (wave == 0) tdm_load_tile(xb + (size_t)b * NSEQ * DDIM, ktaddr);

  for (int jt = 0; jt < NSEQ / 128; ++jt) {
    const int cur = jt & 1;
    if (wave == 0) __builtin_amdgcn_s_wait_tensorcnt(0);
    __syncthreads();                                  // tile 'cur' is ready
    if (wave == 0 && jt + 1 < NSEQ / 128)
      tdm_load_tile(xb + ((size_t)b * NSEQ + (jt + 1) * 128) * DDIM,
                    ktaddr + (unsigned)((1 - cur) * TELEMS * sizeof(__bf16)));
    const __bf16* ktile = kt + cur * TELEMS;

    v8f S[8];
#pragma unroll
    for (int cb = 0; cb < 8; ++cb) {                  // 8 blocks of 16 key cols
      v8f acc = {};
#pragma unroll
      for (int c = 0; c < 4; ++c) {                   // K = 128 in 4x32 chunks
        v16bf bfr = make_frag(ktile + (cb * 16 + lr) * TSTRIDE,
                              32 * c + 8 * h, 32 * c + 16 + 8 * h);
        acc = __builtin_amdgcn_wmma_f32_16x16x32_bf16(
            false, afr[c], false, bfr, (short)0, acc, false, false);
      }
      S[cb] = acc;
    }

#pragma unroll
    for (int v = 0; v < 8; ++v) {
      float tm = S[0][v];
#pragma unroll
      for (int cb = 1; cb < 8; ++cb) tm = fmaxf(tm, S[cb][v]);
#pragma unroll
      for (int d = 1; d < 16; d <<= 1) tm = fmaxf(tm, __shfl_xor(tm, d, 32));
      float nm = fmaxf(m[v], tm);
      float rs = 0.f;
#pragma unroll
      for (int cb = 0; cb < 8; ++cb) rs += exp2f((S[cb][v] - nm) * L2E);
#pragma unroll
      for (int d = 1; d < 16; d <<= 1) rs += __shfl_xor(rs, d, 32);
      z[v] = z[v] * exp2f((m[v] - nm) * L2E) + rs;
      m[v] = nm;
    }
  }

  if (lr == 0) {
#pragma unroll
    for (int v = 0; v < 8; ++v) {
      int row = b * NSEQ + qbase + wave * 16 + v + 8 * h;
      mrow[row] = m[v];
      zinv[row] = 1.0f / z[v];
    }
  }
}

// ---------------------------------------------------------------- kernel 2
// Column sums c_j = sum_i exp(s_ij - m_i)/Z_i, then partial out = sum_j c_j x_j.
__global__ __launch_bounds__(256) void accum_kernel(const __bf16* __restrict__ xb,
                                                    const float* __restrict__ x32,
                                                    const float* __restrict__ mrow,
                                                    const float* __restrict__ zinv,
                                                    float* __restrict__ partial) {
  const int b     = blockIdx.x >> 5;
  const int kti   = blockIdx.x & 31;
  const int jbase = kti * 128;
  const int wave  = threadIdx.x >> 5;
  const int lane  = threadIdx.x & 31;
  const int h     = lane >> 4;
  const int lr    = lane & 15;

  __shared__ __attribute__((aligned(16))) __bf16 qtile[2 * TELEMS];
  __shared__ float ml[2][128], zl[2][128], cl[128], so[128];
  const unsigned qtaddr = (unsigned)(unsigned long long)(&qtile[0]);

  // Fixed B fragments: this wave's 16 keys.
  const __bf16* krow = xb + ((size_t)b * NSEQ + jbase + wave * 16 + lr) * DDIM;
  v16bf bfr[4];
#pragma unroll
  for (int c = 0; c < 4; ++c)
    bfr[c] = make_frag(krow, 32 * c + 8 * h, 32 * c + 16 + 8 * h);

  float csum = 0.f;

  if (wave == 0) tdm_load_tile(xb + (size_t)b * NSEQ * DDIM, qtaddr);

  for (int it = 0; it < NSEQ / 128; ++it) {
    const int cur = it & 1;
    if (threadIdx.x < 128) {                          // stats for this row tile
      ml[cur][threadIdx.x] = mrow[b * NSEQ + it * 128 + threadIdx.x];
      zl[cur][threadIdx.x] = zinv[b * NSEQ + it * 128 + threadIdx.x];
    }
    if (wave == 0) __builtin_amdgcn_s_wait_tensorcnt(0);
    __syncthreads();                                  // tile + stats ready
    if (wave == 0 && it + 1 < NSEQ / 128)
      tdm_load_tile(xb + ((size_t)b * NSEQ + (it + 1) * 128) * DDIM,
                    qtaddr + (unsigned)((1 - cur) * TELEMS * sizeof(__bf16)));
    const __bf16* qtl = qtile + cur * TELEMS;

#pragma unroll
    for (int sb = 0; sb < 8; ++sb) {                  // 8 sub-tiles of 16 rows
      v8f acc = {};
#pragma unroll
      for (int c = 0; c < 4; ++c) {
        v16bf afr = make_frag(qtl + (sb * 16 + lr) * TSTRIDE,
                              32 * c + 8 * h, 32 * c + 16 + 8 * h);
        acc = __builtin_amdgcn_wmma_f32_16x16x32_bf16(
            false, afr, false, bfr[c], (short)0, acc, false, false);
      }
#pragma unroll
      for (int v = 0; v < 8; ++v) {
        int rloc = sb * 16 + v + 8 * h;
        csum += exp2f((acc[v] - ml[cur][rloc]) * L2E) * zl[cur][rloc];
      }
    }
  }

  csum += __shfl_xor(csum, 16, 32);
  if (h == 0) cl[wave * 16 + lr] = csum;
  __syncthreads();

  int d    = threadIdx.x & 127;
  int half = threadIdx.x >> 7;
  float acc = 0.f;
  for (int j = half * 64; j < half * 64 + 64; ++j)
    acc += cl[j] * x32[((size_t)b * NSEQ + jbase + j) * DDIM + d];
  if (half == 0) so[d] = acc;
  __syncthreads();
  if (half == 1) partial[((size_t)b * 32 + kti) * DDIM + d] = so[d] + acc;
}

// ---------------------------------------------------------------- kernel 3
__global__ void reduce_kernel(const float* __restrict__ partial, float* __restrict__ out) {
  int b = blockIdx.x, d = threadIdx.x;
  float s = 0.f;
  for (int jt = 0; jt < 32; ++jt) s += partial[((size_t)b * 32 + jt) * DDIM + d];
  out[b * DDIM + d] = s;
}

// ---------------------------------------------------------------- launch
extern "C" void kernel_launch(void* const* d_in, const int* in_sizes, int n_in,
                              void* d_out, int out_size, void* d_ws, size_t ws_size,
                              hipStream_t stream) {
  (void)in_sizes; (void)n_in; (void)out_size; (void)ws_size;
  const float* x = (const float*)d_in[0];
  float* out = (float*)d_out;

  char* ws = (char*)d_ws;
  __bf16* xb = (__bf16*)ws;
  size_t off = (size_t)BATCH * NSEQ * DDIM * sizeof(__bf16);   // 8 MiB
  float* mrow = (float*)(ws + off); off += (size_t)BATCH * NSEQ * sizeof(float);
  float* zinv = (float*)(ws + off); off += (size_t)BATCH * NSEQ * sizeof(float);
  float* partial = (float*)(ws + off);                          // B*32*128 f32

  const int n = BATCH * NSEQ * DDIM;
  cvt_kernel<<<4096, 256, 0, stream>>>(x, xb, n);
  stats_kernel<<<BATCH * (NSEQ / 128), 256, 0, stream>>>(xb, mrow, zinv);
  accum_kernel<<<BATCH * (NSEQ / 128), 256, 0, stream>>>(xb, x, mrow, zinv, partial);
  reduce_kernel<<<BATCH, DDIM, 0, stream>>>(partial, out);
}
